// LongcatFlashMoE_37752762532032
// MI455X (gfx1250) — compile-verified, matrix operand order checked by reading, not run on hardware
//
#include <hip/hip_runtime.h>
#include <hip/hip_bf16.h>

// ---------------------------------------------------------------------------
// LongcatFlash MoE for gfx1250 (MI455X): dense-over-experts, bf16 WMMA path.
//   T=2048 tokens, H=1024 hidden, I=512 intermediate, E=32 routed + 8 zero.
// Strategy:
//   - one-time fp32->bf16 conversion of all GEMM operands
//   - GEMM1 (hidden @ w13^T) + silu, with combine[t,e] folded into h store
//   - GEMM2 accumulates over ALL experts in one register tile (no reset),
//     adds the zero-expert term in the epilogue.
// Per wave: 32 tokens (2 A frags) x 4 B tiles -> 12 b128 loads per 8 WMMAs.
// ---------------------------------------------------------------------------

#define T_TOK  2048
#define HDIM   1024
#define IDIM   512
#define E_ROUT 32
#define N_TOT  40          // E_ROUTER + ZERO_E
#define TOPK   4
#define RSCALE 2.5f

typedef __attribute__((ext_vector_type(16))) __bf16 v16bf;
typedef __attribute__((ext_vector_type(8)))  __bf16 v8bf;
typedef __attribute__((ext_vector_type(8)))  float  v8f;
typedef __attribute__((ext_vector_type(4)))  float  v4f;

// Workspace layout (bytes), all offsets 16B-aligned
#define OFF_COMBINE 0ull                                          // 256 KB
#define OFF_ZT      (OFF_COMBINE + (size_t)T_TOK * E_ROUT * 4u)   //   8 KB
#define OFF_HIDB    (OFF_ZT + (size_t)T_TOK * 4u)                 //   4 MB
#define OFF_W13B    (OFF_HIDB + (size_t)T_TOK * HDIM * 2u)        //  64 MB
#define OFF_W2B     (OFF_W13B + (size_t)E_ROUT * 2 * IDIM * HDIM * 2u) // 32 MB
#define OFF_HWS     (OFF_W2B + (size_t)E_ROUT * HDIM * IDIM * 2u) //  64 MB
// total ~164.3 MB

// --- WMMA helpers -----------------------------------------------------------
// CDNA5 16-bit A/B fragment: lanes 0-15 hold K {0..7, 16..23}; lanes 16-31
// hold K {8..15, 24..31} (ISA 7.12.2). kh = 0 or 8 selects the half.

__device__ __forceinline__ v16bf load_frag_bf16(const __bf16* p, int kh) {
  v8bf lo = *(const v8bf*)(p + kh);
  v8bf hi = *(const v8bf*)(p + 16 + kh);
  v16bf r;
#pragma unroll
  for (int i = 0; i < 8; ++i) { r[i] = lo[i]; r[i + 8] = hi[i]; }
  return r;
}

__device__ __forceinline__ v8f wmma_bf16(v16bf a, v16bf b, v8f c) {
  // (neg_a, A, neg_b, B, c_mod, C, reuse_a, reuse_b)
  return __builtin_amdgcn_wmma_f32_16x16x32_bf16(false, a, false, b,
                                                 (short)0, c, false, false);
}

// --- Kernel 0: one-time fp32 -> bf16 conversion (vector 8-wide) -------------
__global__ __launch_bounds__(256)
void cvt_f32_bf16(const float* __restrict__ src, __bf16* __restrict__ dst,
                  int n8) {
  int i = blockIdx.x * 256 + threadIdx.x;
  if (i >= n8) return;
  size_t base = (size_t)i * 8;
  v4f a = *(const v4f*)(src + base);
  v4f b = *(const v4f*)(src + base + 4);
  v8bf o;
#pragma unroll
  for (int j = 0; j < 4; ++j) { o[j] = (__bf16)a[j]; o[4 + j] = (__bf16)b[j]; }
  *(v8bf*)(dst + base) = o;
}

// --- Kernel 1: router (softmax + biased top-4) ------------------------------
__global__ __launch_bounds__(64)
void router_kernel(const float* __restrict__ hs, const float* __restrict__ rw,
                   const float* __restrict__ bias, float* __restrict__ combine,
                   float* __restrict__ zero_total) {
  const int t = blockIdx.x;
  const int tid = threadIdx.x;
  __shared__ float s_logit[N_TOT];
  const float* hrow = hs + (size_t)t * HDIM;

  if (tid < N_TOT) {
    const float* wrow = rw + (size_t)tid * HDIM;
    float acc = 0.f;
#pragma unroll 4
    for (int h = 0; h < HDIM; ++h) acc += hrow[h] * wrow[h];
    s_logit[tid] = acc;
  }
  __syncthreads();

  if (tid == 0) {
    float mx = s_logit[0];
    for (int i = 1; i < N_TOT; ++i) mx = fmaxf(mx, s_logit[i]);
    float scores[N_TOT], biased[N_TOT], den = 0.f;
    for (int i = 0; i < N_TOT; ++i) { scores[i] = __expf(s_logit[i] - mx); den += scores[i]; }
    float inv = 1.0f / den;
    for (int i = 0; i < N_TOT; ++i) { scores[i] *= inv; biased[i] = scores[i] + bias[i]; }
    float comb[E_ROUT];
    for (int i = 0; i < E_ROUT; ++i) comb[i] = 0.f;
    float zt = 0.f;
    for (int k = 0; k < TOPK; ++k) {
      int id = 0; float best = biased[0];
      for (int i = 1; i < N_TOT; ++i)
        if (biased[i] > best) { best = biased[i]; id = i; }
      biased[id] = -1e30f;
      float w = scores[id] * RSCALE;
      if (id >= E_ROUT) zt += w; else comb[id] += w;
    }
    for (int i = 0; i < E_ROUT; ++i) combine[(size_t)t * E_ROUT + i] = comb[i];
    zero_total[t] = zt;
  }
}

// --- Kernel 2: h' = combine[t,e] * silu(gate) * up  (bf16 WMMA) -------------
// grid = (I/32, T/128, E); block = 128 (4 waves). Wave w owns token rows
// [t0+32w, t0+32w+32) (2 A frags) and 4 B tiles: gate{n0,n0+16}, up{n0,n0+16}.
// acc[ai*4+j]: ai = A-frag index (tokens +0/+16), j = B tile.
__global__ __launch_bounds__(128)
void gemm1_silu(const __bf16* __restrict__ hidB, const __bf16* __restrict__ w13B,
                const float* __restrict__ combine, __bf16* __restrict__ h_ws) {
  const int e    = blockIdx.z;
  const int t0   = blockIdx.y * 128;
  const int n0   = blockIdx.x * 32;
  const int wave = threadIdx.x >> 5;
  const int lane = threadIdx.x & 31;
  const int lr   = lane & 15;
  const int kh   = (lane < 16) ? 0 : 8;
  const int mhi  = (lane < 16) ? 0 : 8;
  const int m0   = t0 + wave * 32;

  const __bf16* w13e  = w13B + (size_t)e * (2 * IDIM) * HDIM;
  const __bf16* arow0 = hidB + (size_t)(m0 + lr) * HDIM;
  const __bf16* arow1 = hidB + (size_t)(m0 + 16 + lr) * HDIM;
  const __bf16* brow[4];
  brow[0] = w13e + (size_t)(n0 + lr) * HDIM;               // gate n0
  brow[1] = w13e + (size_t)(n0 + 16 + lr) * HDIM;          // gate n0+16
  brow[2] = w13e + (size_t)(IDIM + n0 + lr) * HDIM;        // up   n0
  brow[3] = w13e + (size_t)(IDIM + n0 + 16 + lr) * HDIM;   // up   n0+16

  v8f acc[8];
#pragma unroll
  for (int j = 0; j < 8; ++j)
#pragma unroll
    for (int r = 0; r < 8; ++r) acc[j][r] = 0.f;

  for (int k0 = 0; k0 < HDIM; k0 += 32) {
    __builtin_prefetch(brow[0] + k0 + 256, 0, 1);   // global_prefetch_b8
    v16bf a0 = load_frag_bf16(arow0 + k0, kh);
    v16bf a1 = load_frag_bf16(arow1 + k0, kh);
#pragma unroll
    for (int j = 0; j < 4; ++j) {
      v16bf b = load_frag_bf16(brow[j] + k0, kh);
      acc[j]     = wmma_bf16(a0, b, acc[j]);
      acc[4 + j] = wmma_bf16(a1, b, acc[4 + j]);
    }
  }

  // h' = combine * silu(gate) * up, store bf16. D layout: vgpr r,
  // lanes 0-15 -> M=r, lanes 16-31 -> M=r+8; N = n0 + j*16 + (lane&15).
#pragma unroll
  for (int ai = 0; ai < 2; ++ai) {
#pragma unroll
    for (int j = 0; j < 2; ++j) {
#pragma unroll
      for (int r = 0; r < 8; ++r) {
        float g  = acc[ai * 4 + j][r];
        float u  = acc[ai * 4 + j + 2][r];
        int tok  = m0 + ai * 16 + r + mhi;
        float w  = combine[(size_t)tok * E_ROUT + e];
        float hv = w * (g / (1.0f + __expf(-g))) * u;
        int n    = n0 + j * 16 + lr;
        h_ws[((size_t)e * T_TOK + tok) * IDIM + n] = (__bf16)hv;
      }
    }
  }
}

// --- Kernel 3: out = sum_e (h'[e] @ w2[e]^T) + zt*hidden --------------------
// grid = (H/64, T/128); block = 128 (4 waves). Wave w owns token rows
// [t0+32w, t0+32w+32) and 4 column tiles. Accumulates across ALL experts and
// K in one register tile (combine already folded into h'), no atomics.
__global__ __launch_bounds__(128)
void gemm2_accum(const __bf16* __restrict__ h_ws, const __bf16* __restrict__ w2B,
                 const float* __restrict__ zero_total, const float* __restrict__ hs,
                 float* __restrict__ out) {
  const int t0   = blockIdx.y * 128;
  const int c0   = blockIdx.x * 64;
  const int wave = threadIdx.x >> 5;
  const int lane = threadIdx.x & 31;
  const int lr   = lane & 15;
  const int kh   = (lane < 16) ? 0 : 8;
  const int mhi  = (lane < 16) ? 0 : 8;
  const int m0   = t0 + wave * 32;

  v8f acc[8];
#pragma unroll
  for (int s = 0; s < 8; ++s)
#pragma unroll
    for (int r = 0; r < 8; ++r) acc[s][r] = 0.f;

  for (int e = 0; e < E_ROUT; ++e) {
    const __bf16* arow0 = h_ws + ((size_t)e * T_TOK + (m0 + lr)) * IDIM;
    const __bf16* arow1 = h_ws + ((size_t)e * T_TOK + (m0 + 16 + lr)) * IDIM;
    const __bf16* w2e   = w2B + (size_t)e * HDIM * IDIM;

    for (int k0 = 0; k0 < IDIM; k0 += 32) {
      v16bf a0 = load_frag_bf16(arow0 + k0, kh);
      v16bf a1 = load_frag_bf16(arow1 + k0, kh);
#pragma unroll
      for (int s = 0; s < 4; ++s) {
        const __bf16* br = w2e + (size_t)(c0 + s * 16 + lr) * IDIM + k0;
        v16bf b = load_frag_bf16(br, kh);
        acc[s]     = wmma_bf16(a0, b, acc[s]);
        acc[4 + s] = wmma_bf16(a1, b, acc[4 + s]);
      }
    }
  }

  // epilogue: add zero-expert contribution, single store per element
#pragma unroll
  for (int ai = 0; ai < 2; ++ai) {
#pragma unroll
    for (int r = 0; r < 8; ++r) {
      int tok  = m0 + ai * 16 + r + mhi;
      float zt = zero_total[tok];
#pragma unroll
      for (int s = 0; s < 4; ++s) {
        int col = c0 + s * 16 + lr;
        out[(size_t)tok * HDIM + col] =
            acc[ai * 4 + s][r] + zt * hs[(size_t)tok * HDIM + col];
      }
    }
  }
}

// ---------------------------------------------------------------------------
extern "C" void kernel_launch(void* const* d_in, const int* in_sizes, int n_in,
                              void* d_out, int out_size, void* d_ws, size_t ws_size,
                              hipStream_t stream) {
  (void)in_sizes; (void)n_in; (void)out_size; (void)ws_size;
  const float* hs   = (const float*)d_in[0];
  const float* rw   = (const float*)d_in[1];
  const float* bias = (const float*)d_in[2];
  const float* w13  = (const float*)d_in[3];
  const float* w2   = (const float*)d_in[4];
  float* out = (float*)d_out;

  char* ws = (char*)d_ws;
  float*  combine = (float*)(ws + OFF_COMBINE);
  float*  zt      = (float*)(ws + OFF_ZT);
  __bf16* hidB    = (__bf16*)(ws + OFF_HIDB);
  __bf16* w13B    = (__bf16*)(ws + OFF_W13B);
  __bf16* w2B     = (__bf16*)(ws + OFF_W2B);
  __bf16* hws     = (__bf16*)(ws + OFF_HWS);

  // One-time operand conversion to bf16 (vector-8 per thread)
  {
    int n8_hid = T_TOK * HDIM / 8;                 // 262144
    int n8_w13 = E_ROUT * 2 * IDIM * HDIM / 8;     // 4194304
    int n8_w2  = E_ROUT * HDIM * IDIM / 8;         // 2097152
    cvt_f32_bf16<<<(n8_hid + 255) / 256, 256, 0, stream>>>(hs, hidB, n8_hid);
    cvt_f32_bf16<<<(n8_w13 + 255) / 256, 256, 0, stream>>>(w13, w13B, n8_w13);
    cvt_f32_bf16<<<(n8_w2 + 255) / 256, 256, 0, stream>>>(w2, w2B, n8_w2);
  }

  router_kernel<<<T_TOK, 64, 0, stream>>>(hs, rw, bias, combine, zt);
  gemm1_silu<<<dim3(IDIM / 32, T_TOK / 128, E_ROUT), 128, 0, stream>>>(hidB, w13B, combine, hws);
  gemm2_accum<<<dim3(HDIM / 64, T_TOK / 128), 128, 0, stream>>>(hws, w2B, zt, hs, out);
}
